// InferenceModule_6640019440396
// MI455X (gfx1250) — matrix operand order repaired and checked by reading, not compiled, not run on hardware
//
#include <hip/hip_runtime.h>
#include <hip/hip_bf16.h>
#include <math.h>

#define B_ 1024
#define S_ 256
#define H_ 512
#define E_ 64
#define R_ 32
#define L_ 9
#define EPSF 1e-6f

typedef __attribute__((ext_vector_type(16))) __bf16 v16bf;
typedef __attribute__((ext_vector_type(8)))  __bf16 v8bf;
typedef __attribute__((ext_vector_type(8)))  float  v8f;
typedef __attribute__((ext_vector_type(4)))  float  v4f;

// ---------------------------------------------------------------------------
// K0: convert x -> bf16 row-major, and the 4 used W1 slices (eW1[0], rW1[0..2])
//     -> bf16 TRANSPOSED [mod][hcol][s] so WMMA B-fragments are contiguous.
// ---------------------------------------------------------------------------
__global__ void k0_convert(const float* __restrict__ x,
                           const float* __restrict__ eW1,
                           const float* __restrict__ rW1,
                           unsigned short* __restrict__ xb_u,
                           unsigned short* __restrict__ w1bT_u) {
  __bf16* xb   = reinterpret_cast<__bf16*>(xb_u);
  __bf16* w1bT = reinterpret_cast<__bf16*>(w1bT_u);
  const int NX = B_ * S_;
  const int NW = 4 * H_ * S_;
  int tid = blockIdx.x * blockDim.x + threadIdx.x;
  if (tid < NX) {
    xb[tid] = (__bf16)x[tid];
  } else if (tid < NX + NW) {
    int j    = tid - NX;
    int mod  = j / (H_ * S_);
    int rem  = j - mod * (H_ * S_);
    int hcol = rem / S_;
    int s    = rem - hcol * S_;
    float v = (mod == 0) ? eW1[s * H_ + hcol]
                         : rW1[(size_t)(mod - 1) * S_ * H_ + s * H_ + hcol];
    w1bT[j] = (__bf16)v;
  }
}

// ---------------------------------------------------------------------------
// K1: hidden layer h[mod] = tanh(x @ W1[mod] + b1[mod]), bf16 WMMA, f32 accum.
// One wave (32 threads) per 16x16 output tile; K = 256 via 8x wmma 16x16x32.
// A (16x32 bf16) lane layout: lane<16 holds K {k0..k0+15},
// lane>=16 holds K {k0+16..k0+31}; M = lane&15.
// B (32x16 bf16): col = lane&15, element j = B[k0 + (lane>>4)*16 + j, col];
// contiguous in k thanks to the transposed W1 layout.
// ---------------------------------------------------------------------------
__global__ void __launch_bounds__(32)
k1_hidden_wmma(const unsigned short* __restrict__ xb_u,
               const unsigned short* __restrict__ w1bT_u,
               const float* __restrict__ eb1,
               const float* __restrict__ rb1,
               float* __restrict__ h) {
  const __bf16* xb   = reinterpret_cast<const __bf16*>(xb_u);
  const __bf16* w1bT = reinterpret_cast<const __bf16*>(w1bT_u);

  const int ct   = blockIdx.x;   // 0..31  (H/16 col tiles)
  const int rt   = blockIdx.y;   // 0..63  (B/16 row tiles)
  const int mod  = blockIdx.z;   // 0..3
  const int lane = threadIdx.x;  // 0..31 (wave32)
  const int half = lane >> 4;
  const int i16  = lane & 15;

  const __bf16* Arow = xb   + (size_t)(rt * 16 + i16) * S_;
  const __bf16* Bcol = w1bT + (size_t)mod * H_ * S_ + (size_t)(ct * 16 + i16) * S_;

  v8f acc = {};
#pragma unroll
  for (int k0 = 0; k0 < S_; k0 += 32) {
    v8bf a_lo = *(const v8bf*)(Arow + k0 + half * 16);
    v8bf a_hi = *(const v8bf*)(Arow + k0 + half * 16 + 8);
    v8bf b_lo = *(const v8bf*)(Bcol + k0 + half * 16);
    v8bf b_hi = *(const v8bf*)(Bcol + k0 + half * 16 + 8);
    v16bf a, b;
#pragma unroll
    for (int i = 0; i < 8; ++i) {
      a[i] = a_lo[i]; a[i + 8] = a_hi[i];
      b[i] = b_lo[i]; b[i + 8] = b_hi[i];
    }
    acc = __builtin_amdgcn_wmma_f32_16x16x32_bf16(
        /*neg_a=*/false, a, /*neg_b=*/false, b,
        /*c_mod=*/(short)0, acc, /*reuse_a=*/false, /*reuse_b=*/false);
  }

  const float* bias = (mod == 0) ? eb1 : (rb1 + (mod - 1) * H_);
  const int col = ct * 16 + i16;
  const float bcol = bias[col];
  float* Hout = h + (size_t)mod * B_ * H_;
  // C/D layout: lane's N = lane&15; VGPR v -> M = v + 8*(lane>>4)
#pragma unroll
  for (int v = 0; v < 8; ++v) {
    int row = rt * 16 + v + 8 * half;
    Hout[(size_t)row * H_ + col] = tanhf(acc[v] + bcol);
  }
}

// ---------------------------------------------------------------------------
// K2: second MLP layer. mod 0 -> e0[b,64]; mods 1..3 -> r[k][b,32].
// One block per (b, mod); hidden row staged in LDS, coalesced W2 columns.
// ---------------------------------------------------------------------------
__global__ void __launch_bounds__(64)
k2_proj(const float* __restrict__ h,
        const float* __restrict__ eW2, const float* __restrict__ eb2,
        const float* __restrict__ rW2, const float* __restrict__ rb2,
        float* __restrict__ e0, float* __restrict__ rvec) {
  const int b   = blockIdx.x;
  const int mod = blockIdx.y;
  const int tid = threadIdx.x;  // 0..63
  __shared__ float hrow[H_];
  const float* hsrc = h + (size_t)mod * B_ * H_ + (size_t)b * H_;
  for (int i = tid; i < H_; i += 64) hrow[i] = hsrc[i];
  __syncthreads();

  if (mod == 0) {
    float acc = eb2[tid];               // module 0 of eb2
    const float* W = eW2;               // module 0 of eW2: [H_, E_]
#pragma unroll 8
    for (int s = 0; s < H_; ++s) acc += hrow[s] * W[s * E_ + tid];
    e0[(size_t)b * E_ + tid] = acc;
  } else if (tid < R_) {
    const int k = mod - 1;
    float acc = rb2[k * R_ + tid];
    const float* W = rW2 + (size_t)k * H_ * R_;
#pragma unroll 8
    for (int s = 0; s < H_; ++s) acc += hrow[s] * W[s * R_ + tid];
    rvec[((size_t)k * B_ + b)*R_ + tid] = acc;
  }
}

// ---------------------------------------------------------------------------
// K3: the memory-bound core. One block per batch b, 256 threads.
// Stream tpr[b] (512 KB) ONCE with non-temporal float4 loads, accumulating
// M_k[e,f] = sum_r r_k[r] * tpr[b,e,r,f] for k=0..2 into LDS (48 KB).
// Then run the sequential LN chain and the final (i1+i2+i3) @ Z locally.
// ---------------------------------------------------------------------------
__global__ void __launch_bounds__(256)
k3_core(const float* __restrict__ tpr,
        const float* __restrict__ e0,
        const float* __restrict__ rvec,
        const float* __restrict__ ln_g, const float* __restrict__ ln_b,
        const float* __restrict__ Z,
        float* __restrict__ out) {
  const int b   = blockIdx.x;
  const int tid = threadIdx.x;

  __shared__ __align__(16) float M[3][E_][E_];   // 48 KB
  __shared__ float sr[3][R_];
  __shared__ float s_in[E_];
  __shared__ float s_vec[E_];
  __shared__ float s_sum[E_];

  if (tid < 96) {
    int k = tid >> 5, rr = tid & 31;
    sr[k][rr] = rvec[((size_t)k * B_ + b) * R_ + rr];
  }
  if (tid < E_) { s_in[tid] = e0[(size_t)b * E_ + tid]; s_sum[tid] = 0.f; }
  __syncthreads();

  // Streaming pass: thread handles column group f4 (= tid&15) for 4 e-rows.
  const int f4   = tid & 15;
  const int esub = tid >> 4;       // 0..15
  const float* tb = tpr + (size_t)b * (E_ * R_ * E_) + f4 * 4;

  v4f a[4][3];
#pragma unroll
  for (int s = 0; s < 4; ++s)
#pragma unroll
    for (int k = 0; k < 3; ++k) a[s][k] = (v4f){0.f, 0.f, 0.f, 0.f};

#pragma unroll 4
  for (int rr = 0; rr < R_; ++rr) {
    const float w0 = sr[0][rr], w1 = sr[1][rr], w2 = sr[2][rr];
#pragma unroll
    for (int slab = 0; slab < 4; ++slab) {
      const int e = slab * 16 + esub;
      const v4f* p = (const v4f*)(tb + (size_t)e * (R_ * E_) + rr * E_);
      v4f v = __builtin_nontemporal_load(p);
      a[slab][0] += w0 * v;
      a[slab][1] += w1 * v;
      a[slab][2] += w2 * v;
    }
  }
#pragma unroll
  for (int slab = 0; slab < 4; ++slab) {
    const int e = slab * 16 + esub;
#pragma unroll
    for (int k = 0; k < 3; ++k)
      *(v4f*)&M[k][e][f4 * 4] = a[slab][k];
  }
  __syncthreads();

  // Sequential chain: i_k = LN(in . M_k); accumulate step sum.
  for (int k = 0; k < 3; ++k) {
    float t = 0.f;
    if (tid < E_) {
#pragma unroll 8
      for (int e = 0; e < E_; ++e) t += s_in[e] * M[k][e][tid];
      s_vec[tid] = t;
    }
    __syncthreads();
    float o = 0.f;
    if (tid < E_) {
      float sum = 0.f, sq = 0.f;
#pragma unroll 8
      for (int f = 0; f < E_; ++f) { float v = s_vec[f]; sum += v; sq += v * v; }
      const float mu  = sum * (1.f / E_);
      const float var = sq * (1.f / E_) - mu * mu;
      o = ln_g[k * E_ + tid] * (t - mu) * rsqrtf(var + EPSF) + ln_b[k * E_ + tid];
    }
    __syncthreads();                       // all reads of s_in/s_vec done
    if (tid < E_) { s_in[tid] = o; s_sum[tid] += o; }
    __syncthreads();
  }

  if (tid < L_) {
    float acc = 0.f;
#pragma unroll 8
    for (int f = 0; f < E_; ++f) acc += s_sum[f] * Z[f * L_ + tid];
    out[(size_t)b * L_ + tid] = acc;
  }
}

// ---------------------------------------------------------------------------
extern "C" void kernel_launch(void* const* d_in, const int* in_sizes, int n_in,
                              void* d_out, int out_size, void* d_ws, size_t ws_size,
                              hipStream_t stream) {
  const float* x    = (const float*)d_in[0];
  const float* tpr  = (const float*)d_in[1];
  const float* eW1  = (const float*)d_in[2];
  const float* eb1  = (const float*)d_in[3];
  const float* eW2  = (const float*)d_in[4];
  const float* eb2  = (const float*)d_in[5];
  const float* rW1  = (const float*)d_in[6];
  const float* rb1  = (const float*)d_in[7];
  const float* rW2  = (const float*)d_in[8];
  const float* rb2  = (const float*)d_in[9];
  const float* ln_g = (const float*)d_in[10];
  const float* ln_b = (const float*)d_in[11];
  const float* Z    = (const float*)d_in[12];
  float* out = (float*)d_out;

  // Workspace layout (bytes)
  char* ws = (char*)d_ws;
  unsigned short* xb   = (unsigned short*)(ws);              // 1024*256 bf16   = 512 KB
  unsigned short* w1bT = (unsigned short*)(ws + (size_t)524288);   // 4*512*256 bf16 = 1 MB
  float* h   = (float*)(ws + (size_t)1572864);               // 4*1024*512 f32 = 8 MB
  float* e0  = (float*)(ws + (size_t)9961472);               // 1024*64 f32
  float* rv  = (float*)(ws + (size_t)10223616);              // 3*1024*32 f32
  (void)in_sizes; (void)n_in; (void)out_size; (void)ws_size;

  // K0: bf16 convert + W1 transpose  (786432 work items)
  k0_convert<<<3072, 256, 0, stream>>>(x, eW1, rW1, xb, w1bT);
  // K1: bf16 WMMA hidden GEMMs + tanh
  k1_hidden_wmma<<<dim3(H_ / 16, B_ / 16, 4), 32, 0, stream>>>(xb, w1bT, eb1, rb1, h);
  // K2: projections to e0 / r
  k2_proj<<<dim3(B_, 4), 64, 0, stream>>>(h, eW2, eb2, rW2, rb2, e0, rv);
  // K3: single-pass tpr contraction + LN chain + final Z
  k3_core<<<B_, 256, 0, stream>>>(tpr, e0, rv, ln_g, ln_b, Z, out);
}
